// GMAN_29420525978036
// MI455X (gfx1250) — matrix operand order repaired and checked by reading, not compiled, not run on hardware
//
// MI455X (gfx1250) forward pass for the spatiotemporal transformer reference.
//
// Roofline: ~280 GFLOP total, dominated by dense GEMMs (QKV/proj/FFN/conv-as-GEMM)
// -> all routed through v_wmma_f32_16x16x32_f16 (f16 inputs, f32 accumulate).
// Activations (50-200MB) are largely L2-resident (192MB L2); HBM traffic per pass
// is a few hundred MB (~20-30us at 23.3TB/s), so WMMA math and LDS-staged
// attention are the right levers. wave32 everywhere; 128-thread (4-wave) blocks.
//
// Input flattening assumption: harness flattens setup_inputs() with JAX pytree
// semantics (dict keys sorted alphabetically, lists in order):
//   top: 'params' first, then 'x'.
//   params: blocks(3x18), conv(4x{b,w}), conv_norm{be,g}, dec(21), in{W,b,be,g},
//           spatial_embed; then x. Index map derived below.
//
// Workspace: ~380MB used (see offsets). d_ws assumed >= 380MB.

#include <hip/hip_runtime.h>

typedef __attribute__((ext_vector_type(16))) _Float16 v16h;
typedef __attribute__((ext_vector_type(8)))  _Float16 v8h;
typedef __attribute__((ext_vector_type(8)))  float    v8f;

#define WMMA_F16(a, b, c) \
  __builtin_amdgcn_wmma_f32_16x16x32_f16(false, (a), false, (b), (short)0, (c), false, false)

__device__ __forceinline__ v8f vzero8() {
  v8f z;
#pragma unroll
  for (int i = 0; i < 8; ++i) z[i] = 0.f;
  return z;
}

// A-fragment loader: 16(M) x 32(K) f16, per CDNA5 ISA layout:
// lane l(0-15): M=l, f[0..7]=K(hi*8..hi*8+7), f[8..15]=K(16+hi*8..), hi=lane>>4.
__device__ __forceinline__ v16h ldA_h(const _Float16* __restrict__ p, int ld, int l, int hi) {
  const _Float16* r = p + (long)l * ld + hi * 8;
  v8h c0 = *(const v8h*)r;
  v8h c1 = *(const v8h*)(r + 16);
  v16h f;
#pragma unroll
  for (int i = 0; i < 8; ++i) { f[i] = c0[i]; f[8 + i] = c1[i]; }
  return f;
}

__device__ __forceinline__ v16h ldA_f(const float* __restrict__ p, int ld, int l, int hi) {
  const float* r = p + (long)l * ld + hi * 8;
  v16h f;
#pragma unroll
  for (int i = 0; i < 8; ++i) f[i] = (_Float16)r[i];
#pragma unroll
  for (int i = 0; i < 8; ++i) f[8 + i] = (_Float16)r[16 + i];
  return f;
}

// B-fragment loader: 32(K) x 16(N) f16. lane l: column n=l%16,
// f[0..7]=K(hi*8..hi*8+7), f[8..15]=K(16+hi*8..), hi=lane>>4 (matches A).
// p points at element [n0, k0] of a 16(N-major) x 32(K) row-major tile.
__device__ __forceinline__ v16h ldB_h(const _Float16* __restrict__ p, int ld, int l, int hi) {
  const _Float16* r = p + (long)l * ld + hi * 8;
  v8h c0 = *(const v8h*)r;
  v8h c1 = *(const v8h*)(r + 16);
  v16h f;
#pragma unroll
  for (int i = 0; i < 8; ++i) { f[i] = c0[i]; f[8 + i] = c1[i]; }
  return f;
}

// ---------------------------------------------------------------------------
// Generic GEMM: C[M,N] = A[M,K] @ W[N,K]^T + bias[N], optional exact-GELU.
// Wave tile: 32(M) x 64(N). Templated on A dtype + epilogue so the inner loop
// is straight-line (no per-iteration uniform branches).
// ---------------------------------------------------------------------------
template <bool AHALF, bool GELU>
__global__ __launch_bounds__(128) void k_gemm(const void* __restrict__ Ap,
                                              const _Float16* __restrict__ W,
                                              const float* __restrict__ bias,
                                              float* __restrict__ C,
                                              int M, int N, int K) {
  const int wave = blockIdx.x * 4 + (threadIdx.x >> 5);
  const int nt64 = N >> 6;
  const int mt = wave / nt64;
  const int nt = wave - mt * nt64;
  if (mt * 32 >= M) return;
  const int lane = threadIdx.x & 31;
  const int l = lane & 15, hi = lane >> 4;

  v8f acc[2][4];
#pragma unroll
  for (int i = 0; i < 2; ++i)
#pragma unroll
    for (int j = 0; j < 4; ++j) acc[i][j] = vzero8();

  const long a0r = (long)(mt * 32) * K;
  const long a1r = (long)(mt * 32 + 16) * K;
  const long wr  = (long)(nt * 64) * K;
  const float* Af = (const float*)Ap;
  const _Float16* Ah = (const _Float16*)Ap;

  for (int kk = 0; kk < K; kk += 32) {
    v16h a0, a1;
    if (AHALF) {
      a0 = ldA_h(Ah + a0r + kk, K, l, hi);
      a1 = ldA_h(Ah + a1r + kk, K, l, hi);
    } else {
      a0 = ldA_f(Af + a0r + kk, K, l, hi);
      a1 = ldA_f(Af + a1r + kk, K, l, hi);
    }
#pragma unroll
    for (int j = 0; j < 4; ++j) {
      v16h b = ldB_h(W + wr + (long)(j * 16) * K + kk, K, l, hi);
      acc[0][j] = WMMA_F16(a0, b, acc[0][j]);
      acc[1][j] = WMMA_F16(a1, b, acc[1][j]);
    }
  }
  // C/D layout: vgpr v -> row (hi*8+v) within 16-tile, col = lane&15.
#pragma unroll
  for (int i = 0; i < 2; ++i) {
    const int mrow = mt * 32 + i * 16 + hi * 8;
#pragma unroll
    for (int j = 0; j < 4; ++j) {
      const int col = nt * 64 + j * 16 + l;
      const float bv = bias[col];
#pragma unroll
      for (int v = 0; v < 8; ++v) {
        float x = acc[i][j][v] + bv;
        if (GELU) x = 0.5f * x * (1.0f + erff(x * 0.7071067811865475f));
        C[(long)(mrow + v) * N + col] = x;
      }
    }
  }
}

// ---------------------------------------------------------------------------
// Fused input: h = gelu(ln(x @ Win^T + b)) + spatial_embed.  One block/token.
// ---------------------------------------------------------------------------
__global__ __launch_bounds__(128) void k_input(const float* __restrict__ x,
                                               const float* __restrict__ Wi,
                                               const float* __restrict__ bi,
                                               const float* __restrict__ g,
                                               const float* __restrict__ be,
                                               const float* __restrict__ se,
                                               float* __restrict__ h) {
  __shared__ float sb[4];
  const int r = blockIdx.x, c = threadIdx.x;
  const int n = (r / 24) & 255;
  const float x0 = x[(long)r * 2], x1 = x[(long)r * 2 + 1];
  float v = Wi[c * 2] * x0 + Wi[c * 2 + 1] * x1 + bi[c];
  float sm = v;
#pragma unroll
  for (int o = 16; o; o >>= 1) sm += __shfl_xor(sm, o, 32);
  if ((c & 31) == 0) sb[c >> 5] = sm;
  __syncthreads();
  const float mean = (sb[0] + sb[1] + sb[2] + sb[3]) * (1.f / 128.f);
  __syncthreads();
  const float d = v - mean;
  float s2 = d * d;
#pragma unroll
  for (int o = 16; o; o >>= 1) s2 += __shfl_xor(s2, o, 32);
  if ((c & 31) == 0) sb[c >> 5] = s2;
  __syncthreads();
  const float var = (sb[0] + sb[1] + sb[2] + sb[3]) * (1.f / 128.f);
  float u = d * rsqrtf(var + 1e-5f) * g[c] + be[c];
  u = 0.5f * u * (1.0f + erff(u * 0.7071067811865475f));
  h[(long)r * 128 + c] = u + se[n * 128 + c];
}

// LayerNorm over last dim 128. mode 0: identity rows. mode 1: output row r in
// (b,t,n) order reads input row (b*256+n)*24+t (for the spatial-attn path).
__global__ __launch_bounds__(128) void k_ln(const float* __restrict__ x, float* __restrict__ y,
                                            const float* __restrict__ g,
                                            const float* __restrict__ be, int mode) {
  __shared__ float sb[4];
  const int r = blockIdx.x, c = threadIdx.x;
  long ir = r;
  if (mode == 1) {
    const int b = r / 6144;
    const int rem = r - b * 6144;
    const int t = rem >> 8, n = rem & 255;
    ir = (long)(b * 256 + n) * 24 + t;
  }
  const float v = x[ir * 128 + c];
  float sm = v;
#pragma unroll
  for (int o = 16; o; o >>= 1) sm += __shfl_xor(sm, o, 32);
  if ((c & 31) == 0) sb[c >> 5] = sm;
  __syncthreads();
  const float mean = (sb[0] + sb[1] + sb[2] + sb[3]) * (1.f / 128.f);
  __syncthreads();
  const float d = v - mean;
  float s2 = d * d;
#pragma unroll
  for (int o = 16; o; o >>= 1) s2 += __shfl_xor(s2, o, 32);
  if ((c & 31) == 0) sb[c >> 5] = s2;
  __syncthreads();
  const float var = (sb[0] + sb[1] + sb[2] + sb[3]) * (1.f / 128.f);
  y[(long)r * 128 + c] = d * rsqrtf(var + 1e-5f) * g[c] + be[c];
}

// im2col for causal dilated conv: A[m, k*128+ic] = h[row(m)-shift, ic] (f16).
__global__ __launch_bounds__(384) void k_im2col(const float* __restrict__ h,
                                                _Float16* __restrict__ A, int dil) {
  const int m = blockIdx.x, j = threadIdx.x;
  const int k = j >> 7, ic = j & 127;
  const int t = m % 24;
  const int shift = (2 - k) * dil;
  float v = 0.f;
  if (t - shift >= 0) v = h[(long)(m - shift) * 128 + ic];
  A[(long)m * 384 + j] = (_Float16)v;
}

// Gated activation + residual: h = tanh(y[:,:128]) * sigmoid(y[:,128:]) + h.
__global__ void k_gate(const float* __restrict__ y, float* __restrict__ h) {
  const int idx = blockIdx.x * 256 + threadIdx.x;  // 98304*128 elements
  const int m = idx >> 7, c = idx & 127;
  const float a = y[(long)m * 256 + c];
  const float gg = y[(long)m * 256 + 128 + c];
  h[(long)idx] = tanhf(a) * (1.f / (1.f + __expf(-gg))) + h[(long)idx];
}

__global__ void k_add(const float* __restrict__ src, float* __restrict__ dst, int n) {
  const int i = blockIdx.x * 256 + threadIdx.x;
  if (i < n) dst[i] += src[i];
}

// h[(b,n,t)] += C[(b,t,n)] (spatial-attn projection back into (b,n,t) layout).
__global__ __launch_bounds__(128) void k_add_tr(const float* __restrict__ Csrc,
                                                float* __restrict__ h) {
  const int r = blockIdx.x, c = threadIdx.x;
  const int b = r / 6144;
  const int rem = r - b * 6144;
  const int t = rem >> 8, n = rem & 255;
  const long hr = (long)(b * 256 + n) * 24 + t;
  h[hr * 128 + c] += Csrc[(long)r * 128 + c];
}

__global__ __launch_bounds__(128) void k_copy_enc(const float* __restrict__ h,
                                                  float* __restrict__ enc) {
  const int s = blockIdx.x, c = threadIdx.x;
  enc[(long)s * 128 + c] = h[((long)s * 24 + 23) * 128 + c];
}

// Decoder cross-attn result (exact: uniform softmax over identical keys):
// q1[s*12+i] = horizon_embed[i] + (vp[s] @ c_wo^T + c_bo)  (co precomputed).
__global__ __launch_bounds__(128) void k_dec_q(const float* __restrict__ hor,
                                               const float* __restrict__ co,
                                               float* __restrict__ q1) {
  const int r = blockIdx.x, c = threadIdx.x;
  q1[(long)r * 128 + c] = hor[(r % 12) * 128 + c] + co[(long)(r / 12) * 128 + c];
}

__global__ void k_cvt(const float* __restrict__ s, _Float16* __restrict__ d, int n) {
  const int i = blockIdx.x * 256 + threadIdx.x;
  if (i < n) d[i] = (_Float16)s[i];
}

// conv weight (256,128,3) -> f16 (256, 384) with K = [tap0|tap1|tap2] x 128ch.
__global__ __launch_bounds__(384) void k_conv_repack(const float* __restrict__ w,
                                                     _Float16* __restrict__ d) {
  const int oc = blockIdx.x, j = threadIdx.x;
  const int k = j >> 7, ic = j & 127;
  d[oc * 384 + j] = (_Float16)w[(oc * 128 + ic) * 3 + k];
}

// Small attention (temporal T=24 / decoder causal T=12): one wave per (seq,head),
// lane = query index; scores fully unrolled so they live in VGPRs.
template <int TT, bool CAUSAL>
__global__ __launch_bounds__(128) void k_attn_small(const float* __restrict__ qkv,
                                                    float* __restrict__ out) {
  __shared__ float Ks[4][TT][16];
  __shared__ float Vs[4][TT][16];
  const int w = threadIdx.x >> 5, lane = threadIdx.x & 31;
  const int wid = blockIdx.x * 4 + w;
  const int s = wid >> 3, hh = wid & 7;
  const long base = (long)s * TT * 384 + hh * 16;
  for (int i = lane; i < TT * 16; i += 32) {
    const int t = i >> 4, d = i & 15;
    Ks[w][t][d] = qkv[base + (long)t * 384 + 128 + d];
    Vs[w][t][d] = qkv[base + (long)t * 384 + 256 + d];
  }
  __syncthreads();
  if (lane < TT) {
    float q[16];
#pragma unroll
    for (int d = 0; d < 16; ++d) q[d] = qkv[base + (long)lane * 384 + d] * 0.25f;
    float sc[TT];
    float mx = -1e30f;
#pragma unroll
    for (int j = 0; j < TT; ++j) {
      if (CAUSAL && j > lane) { sc[j] = 0.f; continue; }
      float acc = 0.f;
#pragma unroll
      for (int d = 0; d < 16; ++d) acc += q[d] * Ks[w][j][d];
      sc[j] = acc;
      mx = fmaxf(mx, acc);
    }
    float sum = 0.f;
#pragma unroll
    for (int j = 0; j < TT; ++j) {
      const float p = (CAUSAL && j > lane) ? 0.f : __expf(sc[j] - mx);
      sc[j] = p;
      sum += p;
    }
    float o[16];
#pragma unroll
    for (int d = 0; d < 16; ++d) o[d] = 0.f;
#pragma unroll
    for (int j = 0; j < TT; ++j)
#pragma unroll
      for (int d = 0; d < 16; ++d) o[d] += sc[j] * Vs[w][j][d];
    const float inv = 1.f / sum;
#pragma unroll
    for (int d = 0; d < 16; ++d)
      out[(long)(s * TT + lane) * 128 + hh * 16 + d] = o[d] * inv;
  }
}

// ---------------------------------------------------------------------------
// Spatial attention: per block one (b*t group, head); seq 256, d=16.
// WMMA flash-style with exact two-pass softmax (pass1 max, pass2 exp + P@V).
// Q (scale folded) & K staged in LDS padded to K=32 (zeros); V staged d-major.
// ---------------------------------------------------------------------------
__global__ __launch_bounds__(128) void k_attn_spatial(const float* __restrict__ qkv,
                                                      float* __restrict__ out) {
  __shared__ __align__(32) _Float16 Qh[256 * 32];
  __shared__ __align__(32) _Float16 Kh[256 * 32];
  __shared__ __align__(32) _Float16 Vt[16 * 256];
  __shared__ __align__(32) _Float16 Pc[4][16 * 32];

  const int g = blockIdx.x >> 3;
  const int hh = blockIdx.x & 7;
  const int tid = threadIdx.x;
  const int wv = tid >> 5;
  const int lane = tid & 31, l = lane & 15, hi = lane >> 4;

  for (int i = tid; i < 256 * 16; i += 128) {
    const int n = i >> 4, d = i & 15;
    const float* base = qkv + (long)(g * 256 + n) * 384 + hh * 16 + d;
    Qh[n * 32 + d] = (_Float16)(base[0] * 0.25f);
    Qh[n * 32 + 16 + d] = (_Float16)0.f;
    Kh[n * 32 + d] = (_Float16)base[128];
    Kh[n * 32 + 16 + d] = (_Float16)0.f;
    Vt[d * 256 + n] = (_Float16)base[256];
  }
  __syncthreads();

  for (int qi = 0; qi < 4; ++qi) {
    const int qt = wv * 4 + qi;
    const v16h aq = ldA_h(Qh + qt * 16 * 32, 32, l, hi);

    // Pass 1: exact row max over all 256 keys.
    float rmax[8];
#pragma unroll
    for (int v = 0; v < 8; ++v) rmax[v] = -1e30f;
    for (int kt = 0; kt < 16; ++kt) {
      v8f s = vzero8();
      s = WMMA_F16(aq, ldB_h(Kh + kt * 16 * 32, 32, l, hi), s);
#pragma unroll
      for (int v = 0; v < 8; ++v) rmax[v] = fmaxf(rmax[v], s[v]);
    }
#pragma unroll
    for (int o = 1; o < 16; o <<= 1)
#pragma unroll
      for (int v = 0; v < 8; ++v) rmax[v] = fmaxf(rmax[v], __shfl_xor(rmax[v], o, 32));

    // Pass 2: recompute scores, exp, stage P chunk (16x32 f16), P @ V.
    v8f oacc = vzero8();
    float rsum[8];
#pragma unroll
    for (int v = 0; v < 8; ++v) rsum[v] = 0.f;

    for (int kc = 0; kc < 8; ++kc) {
      v8f s0 = vzero8(), s1 = vzero8();
      s0 = WMMA_F16(aq, ldB_h(Kh + (kc * 2 + 0) * 16 * 32, 32, l, hi), s0);
      s1 = WMMA_F16(aq, ldB_h(Kh + (kc * 2 + 1) * 16 * 32, 32, l, hi), s1);
      _Float16* pc = Pc[wv];
#pragma unroll
      for (int v = 0; v < 8; ++v) {
        const float p0 = __expf(s0[v] - rmax[v]);
        const float p1 = __expf(s1[v] - rmax[v]);
        rsum[v] += p0 + p1;
        pc[(hi * 8 + v) * 32 + l] = (_Float16)p0;        // C-layout -> A-layout via LDS
        pc[(hi * 8 + v) * 32 + 16 + l] = (_Float16)p1;
      }
      __syncthreads();  // uniform; makes own LDS writes visible before A-frag load
      const v16h ap = ldA_h(pc, 32, l, hi);
      const v16h bv = ldB_h(Vt + kc * 32, 256, l, hi);
      oacc = WMMA_F16(ap, bv, oacc);
      __syncthreads();  // protect Pc reuse next chunk
    }
#pragma unroll
    for (int o = 1; o < 16; o <<= 1)
#pragma unroll
      for (int v = 0; v < 8; ++v) rsum[v] += __shfl_xor(rsum[v], o, 32);
#pragma unroll
    for (int v = 0; v < 8; ++v) {
      const long r = (long)(g * 256 + qt * 16 + hi * 8 + v);
      out[r * 128 + hh * 16 + l] = oacc[v] / rsum[v];
    }
  }
}

__global__ void k_outproj(const float* __restrict__ q, const float* __restrict__ w,
                          const float* __restrict__ b, float* __restrict__ out, int rows) {
  const int idx = blockIdx.x * 256 + threadIdx.x;
  if (idx >= rows * 4) return;
  const int r = idx >> 2, oc = idx & 3;
  float acc = b[oc];
  const float* qr = q + (long)r * 128;
  const float* wr = w + oc * 128;
  for (int c = 0; c < 128; ++c) acc += qr[c] * wr[c];
  out[idx] = acc;
}

// ---------------------------------------------------------------------------
// Host orchestration
// ---------------------------------------------------------------------------
// Workspace byte offsets
#define WS_H ((size_t)4 << 20)      // h: 98304x128 f32 (50.3MB)
#define WS_B ((size_t)56 << 20)     // hn/hs/attn-out scratch (50.3MB)
#define WS_C ((size_t)112 << 20)    // proj/ffn-out scratch (50.3MB)
#define WS_A ((size_t)168 << 20)    // big scratch: qkv / ffn-mid / conv / decoder (201MB)

// f16 weight region offsets (in halfs, at ws offset 0)
#define BW_STRIDE 262144
#define W_TQKV 0
#define W_TPROJ 49152
#define W_SQKV 65536
#define W_SPROJ 114688
#define W_F1 131072
#define W_F2 196608
#define DW 786432
#define D_CWV (DW + 0)
#define D_CWO (DW + 16384)
#define D_SWI (DW + 32768)
#define D_SWO (DW + 81920)
#define D_F1 (DW + 98304)
#define D_F2 (DW + 163840)
#define CW (DW + 229376)

extern "C" void kernel_launch(void* const* d_in, const int* in_sizes, int n_in,
                              void* d_out, int out_size, void* d_ws, size_t ws_size,
                              hipStream_t stream) {
  (void)in_sizes; (void)n_in; (void)out_size; (void)ws_size;
  // Input index map (JAX pytree order: sorted dict keys at every level):
  //  blocks[i] base = i*18: 0 f1b, 1 f1w, 2 f2b, 3 f2w, 4 n1.be, 5 n1.g,
  //    6 n2.be, 7 n2.g, 8 n3.be, 9 n3.g, 10 s_proj_b, 11 s_proj_w,
  //    12 s_qkv_b, 13 s_qkv_w, 14 t_proj_b, 15 t_proj_w, 16 t_qkv_b, 17 t_qkv_w
  //  conv[i]: b = 54+2i, w = 55+2i;  conv_norm: be=62, g=63
  //  dec base 64: c_bi, c_bo, c_wi, c_wo, cn.be, cn.g, f1b, f1w, f2b, f2w,
  //    fn.be, fn.g, horizon(76), out_b(77), out_w(78), s_bi(79), s_bo(80),
  //    s_wi(81), s_wo(82), sn.be(83), sn.g(84)
  //  in: W=85, b=86, be=87, g=88; spatial_embed=89; x=90
  auto F = [&](int i) { return (const float*)d_in[i]; };

  char* ws = (char*)d_ws;
  _Float16* w16 = (_Float16*)ws;
  float* h = (float*)(ws + WS_H);
  float* bb = (float*)(ws + WS_B);
  float* cc = (float*)(ws + WS_C);
  char* aa = ws + WS_A;
  float* abuf = (float*)aa;

  auto gemm_f = [&](const float* A, const _Float16* W, const float* bias, float* C,
                    int M, int N, int K, bool gelu) {
    const int blocks = ((M / 32) * (N / 64) + 3) / 4;
    if (gelu) k_gemm<false, true><<<blocks, 128, 0, stream>>>(A, W, bias, C, M, N, K);
    else      k_gemm<false, false><<<blocks, 128, 0, stream>>>(A, W, bias, C, M, N, K);
  };
  auto gemm_h = [&](const _Float16* A, const _Float16* W, const float* bias, float* C,
                    int M, int N, int K) {
    const int blocks = ((M / 32) * (N / 64) + 3) / 4;
    k_gemm<true, false><<<blocks, 128, 0, stream>>>(A, W, bias, C, M, N, K);
  };
  auto cvtp = [&](const float* src, long off, int n) {
    k_cvt<<<(n + 255) / 256, 256, 0, stream>>>(src, w16 + off, n);
  };

  // ---- weight conversion to f16 (every call; deterministic) ----
  for (int ib = 0; ib < 3; ++ib) {
    const int b0 = ib * 18;
    const long woff = (long)ib * BW_STRIDE;
    cvtp(F(b0 + 17), woff + W_TQKV, 49152);
    cvtp(F(b0 + 15), woff + W_TPROJ, 16384);
    cvtp(F(b0 + 13), woff + W_SQKV, 49152);
    cvtp(F(b0 + 11), woff + W_SPROJ, 16384);
    cvtp(F(b0 + 1), woff + W_F1, 65536);
    cvtp(F(b0 + 3), woff + W_F2, 65536);
  }
  cvtp(F(66) + 256 * 128, D_CWV, 16384);  // c_wi v-rows only (cross-attn shortcut)
  cvtp(F(67), D_CWO, 16384);
  cvtp(F(81), D_SWI, 49152);
  cvtp(F(82), D_SWO, 16384);
  cvtp(F(71), D_F1, 65536);
  cvtp(F(73), D_F2, 65536);
  for (int ic = 0; ic < 4; ++ic)
    k_conv_repack<<<256, 384, 0, stream>>>(F(55 + 2 * ic), w16 + CW + (long)ic * 98304);

  // ---- input embed: h = gelu(ln(x@Win^T+b)) + spatial_embed ----
  k_input<<<98304, 128, 0, stream>>>(F(90), F(85), F(86), F(88), F(87), F(89), h);

  // ---- 4 gated dilated causal convs (im2col K=384 -> WMMA GEMM -> gate) ----
  _Float16* im2col = (_Float16*)aa;
  float* convy = (float*)(aa + ((size_t)80 << 20));
  for (int ic = 0; ic < 4; ++ic) {
    k_im2col<<<98304, 384, 0, stream>>>(h, im2col, 1 << ic);
    gemm_h(im2col, w16 + CW + (long)ic * 98304, F(54 + 2 * ic), convy, 98304, 256, 384);
    k_gate<<<49152, 256, 0, stream>>>(convy, h);
  }
  k_ln<<<98304, 128, 0, stream>>>(h, h, F(63), F(62), 0);  // conv_norm (in-place safe)

  // ---- 3 transformer blocks ----
  for (int ib = 0; ib < 3; ++ib) {
    const int b0 = ib * 18;
    const long woff = (long)ib * BW_STRIDE;
    // temporal attention
    k_ln<<<98304, 128, 0, stream>>>(h, bb, F(b0 + 5), F(b0 + 4), 0);
    gemm_f(bb, w16 + woff + W_TQKV, F(b0 + 16), abuf, 98304, 384, 128, false);
    k_attn_small<24, false><<<8192, 128, 0, stream>>>(abuf, bb);
    gemm_f(bb, w16 + woff + W_TPROJ, F(b0 + 14), cc, 98304, 128, 128, false);
    k_add<<<49152, 256, 0, stream>>>(cc, h, 98304 * 128);
    // spatial attention (btn layout)
    k_ln<<<98304, 128, 0, stream>>>(h, bb, F(b0 + 7), F(b0 + 6), 1);
    gemm_f(bb, w16 + woff + W_SQKV, F(b0 + 12), abuf, 98304, 384, 128, false);
    k_attn_spatial<<<3072, 128, 0, stream>>>(abuf, bb);
    gemm_f(bb, w16 + woff + W_SPROJ, F(b0 + 10), cc, 98304, 128, 128, false);
    k_add_tr<<<98304, 128, 0, stream>>>(cc, h);
    // FFN
    k_ln<<<98304, 128, 0, stream>>>(h, bb, F(b0 + 9), F(b0 + 8), 0);
    gemm_f(bb, w16 + woff + W_F1, F(b0 + 0), abuf, 98304, 512, 128, true);
    gemm_f(abuf, w16 + woff + W_F2, F(b0 + 2), cc, 98304, 128, 512, false);
    k_add<<<49152, 256, 0, stream>>>(cc, h, 98304 * 128);
  }

  // ---- decoder ----
  float* enc = (float*)aa;                                 // 4096x128
  float* vp = (float*)(aa + ((size_t)4 << 20));            // 4096x128
  float* q1 = (float*)(aa + ((size_t)8 << 20));            // 49152x128
  float* dbig = (float*)(aa + ((size_t)40 << 20));         // qkv / ffn-mid
  k_copy_enc<<<4096, 128, 0, stream>>>(h, enc);
  // cross-attn over identical keys == v-projection (exact): vp = enc@wv^T+bv
  gemm_f(enc, w16 + D_CWV, F(64) + 256, vp, 4096, 128, 128, false);
  gemm_f(vp, w16 + D_CWO, F(65), cc, 4096, 128, 128, false);
  k_dec_q<<<49152, 128, 0, stream>>>(F(76), cc, q1);       // q1 = horizon + co
  // causal self-attention (HZ=12)
  k_ln<<<49152, 128, 0, stream>>>(q1, bb, F(84), F(83), 0);
  gemm_f(bb, w16 + D_SWI, F(79), dbig, 49152, 384, 128, false);
  k_attn_small<12, true><<<8192, 128, 0, stream>>>(dbig, bb);
  gemm_f(bb, w16 + D_SWO, F(80), cc, 49152, 128, 128, false);
  k_add<<<24576, 256, 0, stream>>>(cc, q1, 49152 * 128);
  // FFN
  k_ln<<<49152, 128, 0, stream>>>(q1, bb, F(75), F(74), 0);
  gemm_f(bb, w16 + D_F1, F(70), dbig, 49152, 512, 128, true);
  gemm_f(dbig, w16 + D_F2, F(72), cc, 49152, 128, 512, false);
  k_add<<<24576, 256, 0, stream>>>(cc, q1, 49152 * 128);
  // output projection H -> NC=4
  k_outproj<<<768, 256, 0, stream>>>(q1, F(78), F(77), (float*)d_out, 49152);
}
